// VectorQuantizer_3075196584209
// MI455X (gfx1250) — matrix operand order repaired and checked by reading, not compiled, not run on hardware
//
#include <hip/hip_runtime.h>
#include <hip/hip_bf16.h>

// ---------------------------------------------------------------------------
// VQ-VAE vector quantizer for MI455X (gfx1250, wave32, WMMA bf16 path)
//   z:        [32,1024,256] f32    (d_in[0])
//   embedding:[1024,256]    f32    (d_in[1])
//   ema_cs:   [1024]        f32    (d_in[2])
//   ema_w:    [1024,256]    f32    (d_in[3])
// Outputs concatenated in d_out (all f32):
//   quantized [8388608] | loss [1] | indices [32768] | new_cluster_size [1024]
//   | new_ema_w [262144] | new_embedding [262144]
// ---------------------------------------------------------------------------

typedef __attribute__((ext_vector_type(16))) __bf16 v16bf;
typedef __attribute__((ext_vector_type(8)))  float  v8f;

#define N_ROWS   32768          // B*T
#define DIMD     256
#define KCODES   1024
#define DECAY    0.99f
#define ONE_M_D  0.01f
#define EPS      1e-5f
#define CCOST    0.25f

// ---- workspace layout (bytes) ----
#define WS_EBF    0                            // bf16 codebook  [K*D]*2 = 524288
#define WS_E2     524288                       // f32 ||e||^2    [K]*4   = 4096
#define WS_CNT    528384                       // f32 counts     [K]*4   = 4096
#define WS_DW     532480                       // f32 dw         [K*D]*4 = 1048576
#define WS_LOSS   1581056                      // f32 loss accum          4
#define WS_INVS   1581060                      // f32 1/smoothed [K]*4   = 4096

// ---- output offsets (floats) ----
#define OUT_Q     0
#define OUT_LOSS  8388608
#define OUT_IDX   8388609
#define OUT_NCS   8421377
#define OUT_EMAW  8422401
#define OUT_NEMB  8684545

// ---------------------------------------------------------------------------
__global__ void __launch_bounds__(256) vq_zero(float* __restrict__ p, int n) {
    int i = blockIdx.x * 256 + threadIdx.x;
    int stride = gridDim.x * 256;
    for (; i < n; i += stride) p[i] = 0.0f;
}

// one wave per code: f32 -> bf16 codebook + ||e||^2
__global__ void __launch_bounds__(256) vq_prep(const float* __restrict__ emb,
                                               __bf16* __restrict__ ebf,
                                               float* __restrict__ e2) {
    int code = (blockIdx.x * 256 + threadIdx.x) >> 5;
    int lane = threadIdx.x & 31;
    const float* er = emb + (size_t)code * DIMD;
    __bf16*      br = ebf + (size_t)code * DIMD;
    float s = 0.0f;
#pragma unroll
    for (int j = 0; j < 8; ++j) {
        float v = er[lane + 32 * j];
        br[lane + 32 * j] = (__bf16)v;
        s += v * v;
    }
#pragma unroll
    for (int m = 1; m <= 16; m <<= 1) s += __shfl_xor(s, m, 32);
    if (lane == 0) e2[code] = s;
}

// ---------------------------------------------------------------------------
// main: 1 wave = 16 rows of z vs all 1024 codes via v_wmma_f32_16x16x32_bf16
__global__ void __launch_bounds__(256) vq_main(
    const float*  __restrict__ z,        // [N, D] f32
    const float*  __restrict__ emb,      // [K, D] f32 (for exact gather)
    const __bf16* __restrict__ ebf,      // [K, D] bf16
    const float*  __restrict__ e2,       // [K]
    float* __restrict__ out_q,           // [N, D]
    float* __restrict__ out_idx,         // [N] (as float)
    float* __restrict__ counts,          // [K]
    float* __restrict__ dw,              // [K, D]
    float* __restrict__ lossAcc)
{
    const int lane  = threadIdx.x & 31;
    const int wid   = blockIdx.x * 8 + (threadIdx.x >> 5);
    const int m0    = wid * 16;                 // row-tile base
    const int half  = lane >> 4;
    const int l15   = lane & 15;
    const int mrow  = m0 + l15;

    // ---- A fragments: 16x32 bf16 per k-step, 8 steps cover D=256 ----
    // A 16x32 layout (contiguous, matches B): lane m holds row m;
    // VGPR v -> K=16*half+2v  (pairs).
    const float* zrow = z + (size_t)mrow * DIMD;
    v16bf a[8];
#pragma unroll
    for (int s = 0; s < 8; ++s) {
        const int kk = s * 32;
#pragma unroll
        for (int j = 0; j < 8; ++j) {
            const int ko = 16 * half + 2 * j;
            float2 f = *(const float2*)(zrow + kk + ko);
            a[s][2 * j]     = (__bf16)f.x;
            a[s][2 * j + 1] = (__bf16)f.y;
        }
    }

    float bestVal[8];
    int   bestIdx[8];
#pragma unroll
    for (int v = 0; v < 8; ++v) { bestVal[v] = 3.4e38f; bestIdx[v] = 0; }

    // ---- sweep all codes in 16-wide N-tiles ----
    for (int n0 = 0; n0 < KCODES; n0 += 16) {
        const int col = n0 + l15;
        // B 32x16 bf16: lane n holds column n; lanes 0-15: K=0..15,
        // lanes 16-31: K=16..31 -> one contiguous 32B chunk per lane.
        const __bf16* brow = ebf + (size_t)col * DIMD + 16 * half;
        v8f c = {0.f, 0.f, 0.f, 0.f, 0.f, 0.f, 0.f, 0.f};
#pragma unroll
        for (int s = 0; s < 8; ++s) {
            v16bf b = *(const v16bf*)(brow + s * 32);
            c = __builtin_amdgcn_wmma_f32_16x16x32_bf16(
                    false, a[s], false, b, (short)0, c, false, false);
        }
        const float e2c = e2[col];
#pragma unroll
        for (int v = 0; v < 8; ++v) {
            float d = e2c - 2.0f * c[v];       // ||z||^2 constant per row: drop
            if (d < bestVal[v]) { bestVal[v] = d; bestIdx[v] = col; }
        }
    }

    // ---- reduce min across the 16 column-lanes of each half-wave ----
#pragma unroll
    for (int v = 0; v < 8; ++v) {
        float bv = bestVal[v];
        int   bi = bestIdx[v];
#pragma unroll
        for (int m = 1; m <= 8; m <<= 1) {
            float ov = __shfl_xor(bv, m, 32);
            int   oi = __shfl_xor(bi, m, 32);
            if (ov < bv || (ov == bv && oi < bi)) { bv = ov; bi = oi; }
        }
        bestVal[v] = bv; bestIdx[v] = bi;
    }

    // C/D layout: (lane,VGPR v) -> row M = v + 8*half. Write indices.
    if (l15 == 0) {
#pragma unroll
        for (int v = 0; v < 8; ++v)
            out_idx[m0 + half * 8 + v] = (float)bestIdx[v];
    }

    // ---- gather / quantized write / dw & counts scatter / loss ----
    float lossPart = 0.0f;
#pragma unroll
    for (int r = 0; r < 16; ++r) {
        const int bi = __shfl(bestIdx[r & 7], (r >> 3) * 16, 32);
        const float* er  = emb   + (size_t)bi * DIMD;
        const float* zr  = z     + (size_t)(m0 + r) * DIMD;
        float*       qr  = out_q + (size_t)(m0 + r) * DIMD;
        float*       dwr = dw    + (size_t)bi * DIMD;
#pragma unroll
        for (int j = 0; j < 8; ++j) {
            const int d = lane + 32 * j;       // coalesced
            float q  = er[d];
            float zv = zr[d];
            qr[d] = q;                          // quantized_st == quantized fwd
            float df = zv - q;
            lossPart += df * df;
            atomicAdd(dwr + d, zv);             // segment_sum(flat_z)
        }
        if (lane == 0) atomicAdd(counts + bi, 1.0f);
    }
#pragma unroll
    for (int m = 1; m <= 16; m <<= 1) lossPart += __shfl_xor(lossPart, m, 32);
    if (lane == 0) atomicAdd(lossAcc, lossPart);
}

// ---------------------------------------------------------------------------
// smoothing of cluster sizes (needs global sum) + loss finalize. 1 WG.
__global__ void __launch_bounds__(1024) vq_smooth(
    const float* __restrict__ ema_cs, const float* __restrict__ counts,
    const float* __restrict__ lossAcc,
    float* __restrict__ out_ncs, float* __restrict__ out_loss,
    float* __restrict__ inv_s)
{
    __shared__ float red[1024];
    const int k = threadIdx.x;
    const float ncs = ema_cs[k] * DECAY + ONE_M_D * counts[k];
    red[k] = ncs;
    __syncthreads();
    for (int s = 512; s > 0; s >>= 1) {
        if (k < s) red[k] += red[k + s];
        __syncthreads();
    }
    const float n = red[0];
    const float sm = (ncs + EPS) / (n + (float)KCODES * EPS) * n;
    out_ncs[k] = sm;
    inv_s[k]   = 1.0f / sm;
    if (k == 0)
        out_loss[0] = CCOST * lossAcc[0] / (float)((size_t)N_ROWS * DIMD);
}

// EMA weight update + new embedding, fully coalesced over K*D elements
__global__ void __launch_bounds__(256) vq_ema(
    const float* __restrict__ ema_w, const float* __restrict__ dw,
    const float* __restrict__ inv_s,
    float* __restrict__ out_emaw, float* __restrict__ out_nemb)
{
    int e = blockIdx.x * 256 + threadIdx.x;      // grid covers K*D = 262144
    float w = ema_w[e] * DECAY + ONE_M_D * dw[e];
    out_emaw[e] = w;
    out_nemb[e] = w * inv_s[e >> 8];             // e/D -> code id
}

// ---------------------------------------------------------------------------
extern "C" void kernel_launch(void* const* d_in, const int* in_sizes, int n_in,
                              void* d_out, int out_size, void* d_ws, size_t ws_size,
                              hipStream_t stream) {
    const float* z      = (const float*)d_in[0];
    const float* emb    = (const float*)d_in[1];
    const float* ema_cs = (const float*)d_in[2];
    const float* ema_w  = (const float*)d_in[3];

    char* ws = (char*)d_ws;
    __bf16* ebf    = (__bf16*)(ws + WS_EBF);
    float*  e2     = (float*) (ws + WS_E2);
    float*  counts = (float*) (ws + WS_CNT);
    float*  dwacc  = (float*) (ws + WS_DW);
    float*  lossA  = (float*) (ws + WS_LOSS);
    float*  inv_s  = (float*) (ws + WS_INVS);

    float* out   = (float*)d_out;
    float* o_q    = out + OUT_Q;
    float* o_loss = out + OUT_LOSS;
    float* o_idx  = out + OUT_IDX;
    float* o_ncs  = out + OUT_NCS;
    float* o_emaw = out + OUT_EMAW;
    float* o_nemb = out + OUT_NEMB;

    // zero counts + dw + lossAcc (contiguous: 1024 + 262144 + 1 floats)
    vq_zero<<<512, 256, 0, stream>>>(counts, KCODES + KCODES * DIMD + 1);

    // bf16 codebook + ||e||^2   (1024 codes, 1 wave each, 8 waves/block)
    vq_prep<<<KCODES / 8, 256, 0, stream>>>(emb, ebf, e2);

    // main: 2048 waves (16 rows each), 8 waves per block -> 256 blocks
    vq_main<<<N_ROWS / (16 * 8), 256, 0, stream>>>(
        z, emb, ebf, e2, o_q, o_idx, counts, dwacc, lossA);

    // smoothing + loss
    vq_smooth<<<1, 1024, 0, stream>>>(ema_cs, counts, lossA, o_ncs, o_loss, inv_s);

    // EMA update + new embedding (262144 elements)
    vq_ema<<<(KCODES * DIMD) / 256, 256, 0, stream>>>(
        ema_w, dwacc, inv_s, o_emaw, o_nemb);
}
